// PointNet2_38422777430430
// MI455X (gfx1250) — matrix operand order repaired and checked by reading, not compiled, not run on hardware
//
#include <hip/hip_runtime.h>
#include <hip/hip_bf16.h>

// ---------------------------------------------------------------------------
// PointNet++ forward for MI455X (gfx1250, wave32).
// All GEMM-shaped work (per-neighbor MLPs ~22 GFLOP, SA3 MLP ~2.7 GFLOP) goes
// through v_wmma_f32_16x16x32_f16 with f32 accumulation. FPS / radius search
// are latency-bound scalar stages; head is tiny VALU work.
// ---------------------------------------------------------------------------

typedef _Float16 v16h __attribute__((ext_vector_type(16)));
typedef _Float16 h8   __attribute__((ext_vector_type(8)));
typedef float    v8f  __attribute__((ext_vector_type(8)));

#define KNBR 64
#define NEG_BIG 3.0e38f

// ---------------------------------------------------------------------------
// Weight transpose + f16 conversion:  W[k][n] (f32) -> wT[n][kpad] (f16, pad 0)
// B-matrix fragments then load as 16 contiguous halves per lane.
// ---------------------------------------------------------------------------
__global__ void prep_wT_kernel(const float* __restrict__ W, _Float16* __restrict__ wT,
                               int Kin, int Nw, int Kpad) {
  int i = blockIdx.x * 256 + threadIdx.x;
  if (i >= Nw * Kpad) return;
  int n = i / Kpad, k = i - n * Kpad;
  wT[i] = (_Float16)(k < Kin ? W[(long)k * Nw + n] : 0.0f);
}

// ---------------------------------------------------------------------------
// Farthest-point sampling: one block (256 thr) per batch; points LDS-resident.
// Matches reference: idx[s] = incoming `last`, then d=min(d,dist), last=argmax(d)
// with first-occurrence (smallest index) tie-break. Writes sampled positions q.
// ---------------------------------------------------------------------------
__global__ void fps_kernel(const float* __restrict__ posp, int pstride,
                           float* __restrict__ qout, int N, int S) {
  __shared__ float sp[3 * 2048];
  __shared__ float sd[2048];
  __shared__ float swv[8];
  __shared__ int   swi[8];
  __shared__ int   s_last;

  const int tid = threadIdx.x;
  const int b = blockIdx.x;
  for (int i = tid; i < N; i += 256) {
    long g = (long)(b * N + i) * pstride;
    sp[i] = posp[g]; sp[N + i] = posp[g + 1]; sp[2 * N + i] = posp[g + 2];
    sd[i] = NEG_BIG;   // "inf" -> min() replaces on first step
  }
  if (tid == 0) s_last = 0;
  __syncthreads();

  const int lane = tid & 31, wv = tid >> 5;
  for (int s = 0; s < S; ++s) {
    int last = s_last;
    if (tid == 0) {
      long qo = ((long)b * S + s) * 3;
      qout[qo] = sp[last]; qout[qo + 1] = sp[N + last]; qout[qo + 2] = sp[2 * N + last];
    }
    float lx = sp[last], ly = sp[N + last], lz = sp[2 * N + last];
    float bv = -1.0f; int bi = 0;
    for (int i = tid; i < N; i += 256) {
      float dx = sp[i] - lx, dy = sp[N + i] - ly, dz = sp[2 * N + i] - lz;
      float d = dx * dx + dy * dy + dz * dz;
      float od = sd[i];
      d = d < od ? d : od;
      sd[i] = d;
      if (d > bv) { bv = d; bi = i; }   // strict > keeps smallest i per thread
    }
    // wave32 argmax reduce, smaller index wins ties
    for (int o = 16; o > 0; o >>= 1) {
      float ov = __shfl_xor(bv, o);
      int   oi = __shfl_xor(bi, o);
      if (ov > bv || (ov == bv && oi < bi)) { bv = ov; bi = oi; }
    }
    if (lane == 0) { swv[wv] = bv; swi[wv] = bi; }
    __syncthreads();
    if (tid == 0) {
      float best = swv[0]; int besti = swi[0];
      for (int w = 1; w < 8; ++w)
        if (swv[w] > best || (swv[w] == best && swi[w] < besti)) { best = swv[w]; besti = swi[w]; }
      s_last = besti;
    }
    __syncthreads();
  }
}

// ---------------------------------------------------------------------------
// Radius neighbors: one wave per query. Reference picks the K smallest point
// indices within r (top_k of -cand) => in-order ballot compaction is exact.
// ---------------------------------------------------------------------------
__global__ void radius_nbr_kernel(const float* __restrict__ posp, int pstride,
                                  const float* __restrict__ q,
                                  int* __restrict__ nidx, int* __restrict__ cnt,
                                  float r2, int N, int S) {
  const int lane = threadIdx.x & 31, wv = threadIdx.x >> 5;
  const int qid = blockIdx.x * 8 + wv;
  const int b = qid / S;
  const float qx = q[(long)qid * 3 + 0], qy = q[(long)qid * 3 + 1], qz = q[(long)qid * 3 + 2];

  for (int j = lane; j < KNBR; j += 32) nidx[(long)qid * KNBR + j] = 0;

  int c = 0;
  for (int base = 0; base < N && c < KNBR; base += 32) {   // c is wave-uniform
    int i = base + lane;                                   // N % 32 == 0
    long g = (long)(b * N + i) * pstride;
    float dx = posp[g] - qx, dy = posp[g + 1] - qy, dz = posp[g + 2] - qz;
    bool in = (dx * dx + dy * dy + dz * dz) <= r2;
    unsigned m = (unsigned)__ballot(in);
    int p = c + __popc(m & ((1u << lane) - 1u));
    if (in && p < KNBR) nidx[(long)qid * KNBR + p] = i;
    c += __popc(m);
  }
  if (lane == 0) cnt[qid] = c < KNBR ? c : KNBR;
}

// ---------------------------------------------------------------------------
// Fused gather + 2-layer MLP (WMMA) + masked max-pool over K=64 neighbors.
// One block (128 thr = 4 waves) per query; each wave owns one 16-row M-tile.
//   A (16x32 f16): lane L: M=L&15; elem j -> K=j+16*hf (contiguous, same order as B)
//   B from LDS wT[n][k]: 16 contiguous halves at k = kc*32 + 16*hf
//   C/D f32: lane holds column n=L&15; VGPR v holds row v+8*hf
// ---------------------------------------------------------------------------
template <int FD, int DINP, int DH, int DOUT>
__global__ void gather_mlp_max_kernel(
    const float* __restrict__ featp, int fstride,
    const float* __restrict__ posp, int pstride,
    const float* __restrict__ q,
    const int* __restrict__ nidx, const int* __restrict__ cnt,
    const _Float16* __restrict__ w1T, const float* __restrict__ b1,
    const float* __restrict__ g1, const float* __restrict__ be1,
    const _Float16* __restrict__ w2T, const float* __restrict__ b2,
    float* __restrict__ out, int S, int N) {
  __shared__ _Float16 A1[64 * DINP];
  __shared__ _Float16 W1s[DH * DINP];
  __shared__ _Float16 Hs[64 * DH];
  __shared__ _Float16 W2s[DOUT * DH];
  __shared__ float SC[DH], BE[DH], B1[DH];
  __shared__ float outred[4][DOUT];

  const int tid = threadIdx.x;
  const int qid = blockIdx.x;
  const int b = qid / S;

  for (int i = tid; i < DH * DINP; i += 128) W1s[i] = w1T[i];
  for (int i = tid; i < DOUT * DH; i += 128) W2s[i] = w2T[i];
  for (int i = tid; i < DH; i += 128) {
    SC[i] = g1[i] * rsqrtf(1.0f + 1e-5f);
    BE[i] = be1[i];
    B1[i] = b1[i];
  }

  const float qx = q[(long)qid * 3 + 0], qy = q[(long)qid * 3 + 1], qz = q[(long)qid * 3 + 2];
  for (int i = tid; i < 64 * DINP; i += 128) {
    int row = i / DINP, col = i - row * DINP;
    int nb = nidx[(long)qid * KNBR + row];
    long base = (long)(b * N + nb);
    float v = 0.0f;
    if (col < FD) v = featp[base * fstride + col];
    else if (col < FD + 3) {
      int c = col - FD;
      v = posp[base * pstride + c] - (c == 0 ? qx : (c == 1 ? qy : qz));
    }
    A1[i] = (_Float16)v;
  }
  __syncthreads();

  const int wv = tid >> 5, lane = tid & 31;
  const int ml = lane & 15, hf = lane >> 4;
  const int m0 = wv * 16;

  // ---- GEMM1 (64xDINP @ DINPxDH) + BN-fold + ReLU -> Hs (f16) ----
  for (int nt = 0; nt < DH / 16; ++nt) {
    v8f acc = {};
    for (int kc = 0; kc < DINP / 32; ++kc) {
      const _Float16* ap = &A1[(m0 + ml) * DINP + kc * 32 + hf * 16];
      h8 lo = *(const h8*)(ap);
      h8 hi = *(const h8*)(ap + 8);
      v16h a;
#pragma unroll
      for (int j = 0; j < 8; ++j) { a[j] = lo[j]; a[j + 8] = hi[j]; }
      const _Float16* bp = &W1s[(nt * 16 + ml) * DINP + kc * 32 + hf * 16];
      v16h bf = *(const v16h*)bp;
      acc = __builtin_amdgcn_wmma_f32_16x16x32_f16(false, a, false, bf, (short)0, acc, false, false);
    }
    int n = nt * 16 + ml;
    float sc = SC[n], be = BE[n], bb = B1[n];
#pragma unroll
    for (int vv = 0; vv < 8; ++vv) {
      float h = (acc[vv] + bb) * sc + be;
      h = h > 0.0f ? h : 0.0f;
      Hs[(m0 + vv + 8 * hf) * DH + n] = (_Float16)h;
    }
  }
  __syncthreads();

  // ---- GEMM2 (64xDH @ DHxDOUT) + masked max over valid neighbor rows ----
  const int c_valid = cnt[qid];
  for (int nt = 0; nt < DOUT / 16; ++nt) {
    v8f acc = {};
    for (int kc = 0; kc < DH / 32; ++kc) {
      const _Float16* ap = &Hs[(m0 + ml) * DH + kc * 32 + hf * 16];
      h8 lo = *(const h8*)(ap);
      h8 hi = *(const h8*)(ap + 8);
      v16h a;
#pragma unroll
      for (int j = 0; j < 8; ++j) { a[j] = lo[j]; a[j + 8] = hi[j]; }
      const _Float16* bp = &W2s[(nt * 16 + ml) * DH + kc * 32 + hf * 16];
      v16h bf = *(const v16h*)bp;
      acc = __builtin_amdgcn_wmma_f32_16x16x32_f16(false, a, false, bf, (short)0, acc, false, false);
    }
    float mx = -NEG_BIG;
#pragma unroll
    for (int vv = 0; vv < 8; ++vv) {
      int row = m0 + vv + 8 * hf;
      if (row < c_valid) mx = fmaxf(mx, acc[vv]);
    }
    mx = fmaxf(mx, __shfl_xor(mx, 16));   // combine lane halves (same column n)
    if (hf == 0) outred[wv][nt * 16 + ml] = mx;
  }
  __syncthreads();

  for (int n = tid; n < DOUT; n += 128) {
    float m = fmaxf(fmaxf(outred[0][n], outred[1][n]), fmaxf(outred[2][n], outred[3][n]));
    out[(long)qid * DOUT + n] = m + b2[n];   // bias is row-invariant: add after max
  }
}

// ---------------------------------------------------------------------------
// SA3 hidden layer via WMMA: rows = points, [f2(128)|q2(3)] padded to K=160,
// N = 256 hidden. One block (128 thr = 4 waves) per 64-row chunk of a batch.
// A staged in LDS (f16); B fragments loaded directly from prepped global
// w1T[n][160] (80 KB, L2-resident). Output written f16 for the next GEMM.
// ---------------------------------------------------------------------------
__global__ void sa3_hidden_wmma_kernel(const float* __restrict__ f2, const float* __restrict__ q2,
                                       const _Float16* __restrict__ w1T,
                                       const float* __restrict__ b1, const float* __restrict__ g1,
                                       const float* __restrict__ be1,
                                       _Float16* __restrict__ Hh) {
  __shared__ _Float16 A3[64 * 160];   // 20 KB

  const int tid = threadIdx.x;          // 128
  const int b = blockIdx.x >> 2;
  const int br = blockIdx.x & 3;
  const int rbase = b * 256 + br * 64;  // first global point row of this chunk

  for (int i = tid; i < 64 * 160; i += 128) {
    int r = i / 160, c = i - r * 160;
    long p = (long)(rbase + r);
    float v = 0.0f;
    if (c < 128) v = f2[p * 128 + c];
    else if (c < 131) v = q2[p * 3 + (c - 128)];
    A3[i] = (_Float16)v;
  }
  __syncthreads();

  const int wv = tid >> 5, lane = tid & 31;
  const int ml = lane & 15, hf = lane >> 4;
  const int m0 = wv * 16;

  for (int nt = 0; nt < 16; ++nt) {
    v8f acc = {};
    for (int kc = 0; kc < 5; ++kc) {
      const _Float16* ap = &A3[(m0 + ml) * 160 + kc * 32 + hf * 16];
      h8 lo = *(const h8*)(ap);
      h8 hi = *(const h8*)(ap + 8);
      v16h a;
#pragma unroll
      for (int j = 0; j < 8; ++j) { a[j] = lo[j]; a[j + 8] = hi[j]; }
      const _Float16* bp = &w1T[(long)(nt * 16 + ml) * 160 + kc * 32 + hf * 16];
      v16h bf = *(const v16h*)bp;
      acc = __builtin_amdgcn_wmma_f32_16x16x32_f16(false, a, false, bf, (short)0, acc, false, false);
    }
    int n = nt * 16 + ml;
    float sc = g1[n] * rsqrtf(1.0f + 1e-5f), be = be1[n], bb = b1[n];
#pragma unroll
    for (int vv = 0; vv < 8; ++vv) {
      float h = (acc[vv] + bb) * sc + be;
      h = h > 0.0f ? h : 0.0f;
      Hh[(long)(rbase + m0 + vv + 8 * hf) * 256 + n] = (_Float16)h;
    }
  }
}

// ---------------------------------------------------------------------------
// SA3 output layer via WMMA + global max over the 256 points of each batch:
// out[b][n] = max_p (H[b] @ W2)[p][n] + b2[n].
// One block (256 thr = 8 waves) per (batch, 64-column chunk). W2 chunk staged
// once in LDS (32 KB) -> W2 traffic drops from ~4 GB (scalar version) to 8 MB.
// ---------------------------------------------------------------------------
__global__ void sa3_out_wmma_kernel(const _Float16* __restrict__ Hh,
                                    const _Float16* __restrict__ w2T,
                                    const float* __restrict__ b2,
                                    float* __restrict__ g) {
  __shared__ _Float16 W2s[64 * 256];    // 32 KB
  __shared__ float outred[8][64];

  const int tid = threadIdx.x;          // 256
  const int b = blockIdx.x >> 3;
  const int nc = blockIdx.x & 7;        // which 64 output columns

  for (int i = tid; i < 64 * 256; i += 256)
    W2s[i] = w2T[(long)nc * 64 * 256 + i];
  __syncthreads();

  const int wv = tid >> 5, lane = tid & 31;
  const int ml = lane & 15, hf = lane >> 4;

  float wmax[4] = {-NEG_BIG, -NEG_BIG, -NEG_BIG, -NEG_BIG};

  for (int mi = 0; mi < 2; ++mi) {
    const int m0 = (wv * 2 + mi) * 16;
    const _Float16* arow = &Hh[(long)(b * 256 + m0 + ml) * 256];
    __builtin_prefetch(arow, 0, 1);     // gfx1250 global_prefetch path
    for (int nt = 0; nt < 4; ++nt) {
      v8f acc = {};
      for (int kc = 0; kc < 8; ++kc) {
        const _Float16* ap = arow + kc * 32 + hf * 16;
        h8 lo = *(const h8*)(ap);
        h8 hi = *(const h8*)(ap + 8);
        v16h a;
#pragma unroll
        for (int j = 0; j < 8; ++j) { a[j] = lo[j]; a[j + 8] = hi[j]; }
        const _Float16* bp = &W2s[(nt * 16 + ml) * 256 + kc * 32 + hf * 16];
        v16h bf = *(const v16h*)bp;
        acc = __builtin_amdgcn_wmma_f32_16x16x32_f16(false, a, false, bf, (short)0, acc, false, false);
      }
      float mx = -NEG_BIG;
#pragma unroll
      for (int vv = 0; vv < 8; ++vv) mx = fmaxf(mx, acc[vv]);
      mx = fmaxf(mx, __shfl_xor(mx, 16));
      wmax[nt] = fmaxf(wmax[nt], mx);
    }
  }
  if (hf == 0) {
#pragma unroll
    for (int nt = 0; nt < 4; ++nt) outred[wv][nt * 16 + ml] = wmax[nt];
  }
  __syncthreads();
  if (tid < 64) {
    float m = outred[0][tid];
    for (int w = 1; w < 8; ++w) m = fmaxf(m, outred[w][tid]);
    g[(long)b * 512 + nc * 64 + tid] = m + b2[nc * 64 + tid];
  }
}

// Head: relu(g @ W1 + b1) @ W2 + b2   (tiny: kept f32 VALU for accuracy)
__global__ void head_kernel(const float* __restrict__ g, const float* __restrict__ W1,
                            const float* __restrict__ b1, const float* __restrict__ W2,
                            const float* __restrict__ b2, float* __restrict__ out) {
  const int b = blockIdx.x, tid = threadIdx.x;   // block 256
  __shared__ float gv[512];
  __shared__ float hid[256];
  gv[tid] = g[(long)b * 512 + tid];
  gv[tid + 256] = g[(long)b * 512 + tid + 256];
  __syncthreads();
  float a = b1[tid];
  for (int k = 0; k < 512; ++k) a += gv[k] * W1[(long)k * 256 + tid];
  hid[tid] = a > 0.0f ? a : 0.0f;
  __syncthreads();
  float o0 = b2[tid], o1 = b2[tid + 256];
  for (int k = 0; k < 256; ++k) {
    float h = hid[k];
    o0 += h * W2[(long)k * 512 + tid];
    o1 += h * W2[(long)k * 512 + tid + 256];
  }
  out[(long)b * 512 + tid] = o0;
  out[(long)b * 512 + tid + 256] = o1;
}

// ---------------------------------------------------------------------------
extern "C" void kernel_launch(void* const* d_in, const int* in_sizes, int n_in,
                              void* d_out, int out_size, void* d_ws, size_t ws_size,
                              hipStream_t stream) {
  (void)in_sizes; (void)out_size; (void)ws_size;

  // --- resolve the 23 parameter leaves (handles flat-leaf or per-dict-blob) ---
  const float* P[23];
  auto fill_mlp2 = [](const float* base, int din, int dh, int dout, const float** o) {
    o[0] = base; base += (long)din * dh;
    o[1] = base; base += dh;
    o[2] = base; base += dh;
    o[3] = base; base += dh;
    o[4] = base; base += (long)dh * dout;
    o[5] = base;
  };
  if (n_in >= 23) {
    for (int i = 0; i < 23; ++i) P[i] = (const float*)d_in[i];
  } else {
    P[0] = (const float*)d_in[0];
    fill_mlp2((const float*)d_in[1], 6, 32, 64, &P[1]);
    fill_mlp2((const float*)d_in[2], 67, 64, 128, &P[7]);
    fill_mlp2((const float*)d_in[3], 131, 256, 512, &P[13]);
    const float* h = (const float*)d_in[4];
    P[19] = h; h += 512 * 256;
    P[20] = h; h += 256;
    P[21] = h; h += 256 * 512;
    P[22] = h;
  }
  const float* x = P[0];
  const float *s1W1 = P[1], *s1b1 = P[2], *s1g = P[3], *s1be = P[4], *s1W2 = P[5], *s1b2 = P[6];
  const float *s2W1 = P[7], *s2b1 = P[8], *s2g = P[9], *s2be = P[10], *s2W2 = P[11], *s2b2 = P[12];
  const float *s3W1 = P[13], *s3b1 = P[14], *s3g = P[15], *s3be = P[16], *s3W2 = P[17], *s3b2 = P[18];
  const float *hW1 = P[19], *hb1 = P[20], *hW2 = P[21], *hb2 = P[22];

  // --- workspace carve (~28 MB) ---
  char* ws = (char*)d_ws;
  size_t off = 0;
  auto carve = [&](size_t bytes) -> void* {
    void* p = ws + off;
    off = (off + bytes + 255) & ~(size_t)255;
    return p;
  };
  _Float16* w1aT = (_Float16*)carve((size_t)32 * 32 * 2);
  _Float16* w2aT = (_Float16*)carve((size_t)64 * 32 * 2);
  _Float16* w1bT = (_Float16*)carve((size_t)64 * 96 * 2);
  _Float16* w2bT = (_Float16*)carve((size_t)128 * 64 * 2);
  _Float16* w1cT = (_Float16*)carve((size_t)256 * 160 * 2);
  _Float16* w2cT = (_Float16*)carve((size_t)512 * 256 * 2);
  float* q1 = (float*)carve((size_t)32 * 1024 * 3 * 4);
  int* nidx1 = (int*)carve((size_t)32 * 1024 * KNBR * 4);
  int* cnt1 = (int*)carve((size_t)32 * 1024 * 4);
  float* f1 = (float*)carve((size_t)32 * 1024 * 64 * 4);
  float* q2 = (float*)carve((size_t)32 * 256 * 3 * 4);
  int* nidx2 = (int*)carve((size_t)32 * 256 * KNBR * 4);
  int* cnt2 = (int*)carve((size_t)32 * 256 * 4);
  float* f2 = (float*)carve((size_t)32 * 256 * 128 * 4);
  _Float16* Hh = (_Float16*)carve((size_t)32 * 256 * 256 * 2);
  float* gbuf = (float*)carve((size_t)32 * 512 * 4);

  // --- weight prep (f16 transposed, K-padded) ---
  prep_wT_kernel<<<(32 * 32 + 255) / 256, 256, 0, stream>>>(s1W1, w1aT, 6, 32, 32);
  prep_wT_kernel<<<(64 * 32 + 255) / 256, 256, 0, stream>>>(s1W2, w2aT, 32, 64, 32);
  prep_wT_kernel<<<(64 * 96 + 255) / 256, 256, 0, stream>>>(s2W1, w1bT, 67, 64, 96);
  prep_wT_kernel<<<(128 * 64 + 255) / 256, 256, 0, stream>>>(s2W2, w2bT, 64, 128, 64);
  prep_wT_kernel<<<(256 * 160 + 255) / 256, 256, 0, stream>>>(s3W1, w1cT, 131, 256, 160);
  prep_wT_kernel<<<(512 * 256 + 255) / 256, 256, 0, stream>>>(s3W2, w2cT, 256, 512, 256);

  // --- SA1: N=2048 -> S=1024, r=0.2 ---
  fps_kernel<<<32, 256, 0, stream>>>(x, 6, q1, 2048, 1024);
  radius_nbr_kernel<<<(32 * 1024) / 8, 256, 0, stream>>>(x, 6, q1, nidx1, cnt1, 0.04f, 2048, 1024);
  gather_mlp_max_kernel<3, 32, 32, 64><<<32 * 1024, 128, 0, stream>>>(
      x + 3, 6, x, 6, q1, nidx1, cnt1, w1aT, s1b1, s1g, s1be, w2aT, s1b2, f1, 1024, 2048);

  // --- SA2: N=1024 -> S=256, r=0.4 ---
  fps_kernel<<<32, 256, 0, stream>>>(q1, 3, q2, 1024, 256);
  radius_nbr_kernel<<<(32 * 256) / 8, 256, 0, stream>>>(q1, 3, q2, nidx2, cnt2, 0.16f, 1024, 256);
  gather_mlp_max_kernel<64, 96, 64, 128><<<32 * 256, 128, 0, stream>>>(
      f1, 64, q1, 3, q2, nidx2, cnt2, w1bT, s2b1, s2g, s2be, w2bT, s2b2, f2, 256, 1024);

  // --- SA3: per-point MLP (WMMA) + global max, then head ---
  sa3_hidden_wmma_kernel<<<32 * 4, 128, 0, stream>>>(f2, q2, w1cT, s3b1, s3g, s3be, Hh);
  sa3_out_wmma_kernel<<<32 * 8, 256, 0, stream>>>(Hh, w2cT, s3b2, gbuf);
  head_kernel<<<32, 256, 0, stream>>>(gbuf, hW1, hb1, hW2, hb2, (float*)d_out);
}
